// MultiHeadFastAttention_79439715106997
// MI455X (gfx1250) — matrix operand order repaired and checked by reading, not compiled, not run on hardware
//
#include <hip/hip_runtime.h>

// ---------------------------------------------------------------------------
// Performer (FAVOR+) attention for MI455X (gfx1250), wave32 + WMMA f16->f32.
// B=4, N=4096, D=1024, H=16, M=256, DH=64.
// - All WMMA operands staged in LDS with fragment-major layouts so every
//   A/B fragment is two ds_load_b128 per lane.
// - ksum (column sums of kp) computed by WMMA against a constant ones tile
//   appended to the v operand (no scalar reduction in the hot loop).
// ---------------------------------------------------------------------------

#define B_   4
#define N_   4096
#define D_   1024
#define H_   16
#define M_   256
#define DH_  64
#define EPS_ 1e-4f
#define DATA_SCALE 0.3535533905932738f   // 64^-0.25
#define SCALE_     0.0625f               // 256^-0.5

typedef __attribute__((ext_vector_type(16))) _Float16 v16h;
typedef __attribute__((ext_vector_type(8)))  _Float16 v8h;
typedef __attribute__((ext_vector_type(4)))  _Float16 v4h;
typedef __attribute__((ext_vector_type(8)))  float    v8f;

__device__ __forceinline__ v8f wmma_f16f32(v16h a, v16h b, v8f c) {
  return __builtin_amdgcn_wmma_f32_16x16x32_f16(false, a, false, b, (short)0, c,
                                                false, false);
}

// A fragment (16x32 f16): row-major [row][K] source; two 16B contiguous runs.
__device__ __forceinline__ v16h fragA(const _Float16* rowp, int kbase, int hi) {
  const v8h lo = *(const v8h*)(rowp + kbase + hi * 8);
  const v8h hh = *(const v8h*)(rowp + kbase + 16 + hi * 8);
  return __builtin_shufflevector(lo, hh, 0, 1, 2, 3, 4, 5, 6, 7, 8, 9, 10, 11,
                                 12, 13, 14, 15);
}

// B fragment (32x16 f16): col-major [col][K] source; two 16B contiguous runs.
__device__ __forceinline__ v16h fragB(const _Float16* colp, int kbase, int hi) {
  const v8h lo = *(const v8h*)(colp + kbase + hi * 8);
  const v8h hh = *(const v8h*)(colp + kbase + 16 + hi * 8);
  return __builtin_shufflevector(lo, hh, 0, 1, 2, 3, 4, 5, 6, 7, 8, 9, 10, 11,
                                 12, 13, 14, 15);
}

__device__ __forceinline__ v8h cvt8(float4 a, float4 b, float s) {
  v8h h;
  h[0] = (_Float16)(a.x * s); h[1] = (_Float16)(a.y * s);
  h[2] = (_Float16)(a.z * s); h[3] = (_Float16)(a.w * s);
  h[4] = (_Float16)(b.x * s); h[5] = (_Float16)(b.y * s);
  h[6] = (_Float16)(b.z * s); h[7] = (_Float16)(b.w * s);
  return h;
}

// ---------------------------------------------------------------------------
// Phase 1: per (b,h) compute context[256x64] = kp^T @ v and ksum[256]=kp^T@1.
// ---------------------------------------------------------------------------
__global__ __launch_bounds__(256)
void performer_phase1(const float* __restrict__ kg, const float* __restrict__ vg,
                      const float* __restrict__ proj, float* __restrict__ ctx,
                      float* __restrict__ ksum) {
  __shared__ _Float16 projB[M_][DH_ + 8];     // col-major B: [m][d]
  __shared__ _Float16 kh[32][DH_ + 8];        // A: [n][d] (scaled)
  __shared__ _Float16 vT[DH_ + 16][32 + 8];   // col-major B: [e][n]; rows 64..79 = ones
  __shared__ _Float16 kpT[M_][32 + 8];        // A for GEMM2: [m][n]
  __shared__ float    dpl[32][M_ + 4];        // dp tile (pre-exp)
  __shared__ float    dnpart[32][8];
  __shared__ float    rpart[32][8];
  __shared__ float    dnv[32];
  __shared__ float    rmax[32];

  const int bh   = blockIdx.x;
  const int b    = bh / H_;
  const int h    = bh % H_;
  const int tid  = threadIdx.x;
  const int wave = tid >> 5;
  const int lane = tid & 31;
  const int lr   = lane & 15;
  const int hi   = lane >> 4;

  // proj [M][DH] is already col-major-by-m for the B operand: direct copy.
  for (int i = tid; i < (M_ * DH_) / 4; i += 256) {
    const float4 f = *(const float4*)(proj + i * 4);
    v4h h4; h4[0] = (_Float16)f.x; h4[1] = (_Float16)f.y;
    h4[2] = (_Float16)f.z; h4[3] = (_Float16)f.w;
    *(v4h*)&projB[(i * 4) >> 6][(i * 4) & 63] = h4;
  }
  // constant ones tile used to fold ksum into the WMMA pipe
  for (int i = tid; i < 16 * 32; i += 256)
    vT[DH_ + (i >> 5)][i & 31] = (_Float16)1.0f;

  v8f acc[2][5];
  #pragma unroll
  for (int i = 0; i < 2; ++i)
    #pragma unroll
    for (int j = 0; j < 5; ++j)
      #pragma unroll
      for (int vv = 0; vv < 8; ++vv) acc[i][j][vv] = 0.f;

  const int row = tid >> 3;          // 0..31
  const int off = (tid & 7) * 8;     // 0..56

  __syncthreads();

  for (int nt = 0; nt < N_; nt += 32) {
    // ---- load k/v tiles (float4), k scaled->f16, v transposed -------------
    {
      const size_t base = ((size_t)b * N_ + nt + row) * D_ + h * DH_ + off;
      const float4 k0 = *(const float4*)(kg + base);
      const float4 k1 = *(const float4*)(kg + base + 4);
      const float4 w0 = *(const float4*)(vg + base);
      const float4 w1 = *(const float4*)(vg + base + 4);
      const float ss = k0.x * k0.x + k0.y * k0.y + k0.z * k0.z + k0.w * k0.w +
                       k1.x * k1.x + k1.y * k1.y + k1.z * k1.z + k1.w * k1.w;
      dnpart[row][tid & 7] = ss;
      *(v8h*)&kh[row][off] = cvt8(k0, k1, DATA_SCALE);
      vT[off + 0][row] = (_Float16)w0.x; vT[off + 1][row] = (_Float16)w0.y;
      vT[off + 2][row] = (_Float16)w0.z; vT[off + 3][row] = (_Float16)w0.w;
      vT[off + 4][row] = (_Float16)w1.x; vT[off + 5][row] = (_Float16)w1.y;
      vT[off + 6][row] = (_Float16)w1.z; vT[off + 7][row] = (_Float16)w1.w;
    }
    __syncthreads();

    // ---- GEMM1: dp[32x256] = kh @ projB^T (K=64) --------------------------
    #pragma unroll
    for (int i = 0; i < 4; ++i) {
      const int t = i * 8 + wave;          // 2 n-tiles x 16 m-tiles
      const int ntile = t >> 4, mtile = t & 15;
      const _Float16* ap = &kh[ntile * 16 + lr][0];
      const _Float16* bp = &projB[mtile * 16 + lr][0];
      v8f c;
      #pragma unroll
      for (int vv = 0; vv < 8; ++vv) c[vv] = 0.f;
      c = wmma_f16f32(fragA(ap, 0, hi), fragB(bp, 0, hi), c);
      c = wmma_f16f32(fragA(ap, 32, hi), fragB(bp, 32, hi), c);
      #pragma unroll
      for (int vv = 0; vv < 8; ++vv)
        dpl[ntile * 16 + vv + 8 * hi][mtile * 16 + lr] = c[vv];
    }
    __syncthreads();

    // ---- parallel row-max partials ---------------------------------------
    {
      const int r2 = tid >> 3, c0 = (tid & 7) * 32;
      float mx = -3.4e38f;
      #pragma unroll
      for (int j = 0; j < 32; ++j) mx = fmaxf(mx, dpl[r2][c0 + j]);
      rpart[r2][tid & 7] = mx;
    }
    __syncthreads();
    if (tid < 32) {
      float s = 0.f, mx = -3.4e38f;
      #pragma unroll
      for (int j = 0; j < 8; ++j) { s += dnpart[tid][j]; mx = fmaxf(mx, rpart[tid][j]); }
      dnv[tid]  = s * (0.5f * DATA_SCALE * DATA_SCALE);
      rmax[tid] = mx;
    }
    __syncthreads();

    // ---- kp = scale*exp(dp - dn - rowmax + EPS), stored transposed --------
    {
      const int r2 = tid >> 3;
      const int c0 = (tid & 7) * 32;
      const float dsub = dnv[r2] + rmax[r2] - EPS_;
      #pragma unroll
      for (int j = 0; j < 32; ++j)
        kpT[c0 + j][r2] = (_Float16)(SCALE_ * __expf(dpl[r2][c0 + j] - dsub));
    }
    __syncthreads();

    // ---- GEMM2: [context | ksum] += kp^T @ [v | 1]  (K=32) ----------------
    #pragma unroll
    for (int mt = 0; mt < 2; ++mt) {
      const v16h a = fragA(&kpT[wave * 32 + mt * 16 + lr][0], 0, hi);
      #pragma unroll
      for (int et = 0; et < 5; ++et) {
        const v16h bb = fragB(&vT[et * 16 + lr][0], 0, hi);
        acc[mt][et] = wmma_f16f32(a, bb, acc[mt][et]);
      }
    }
    __syncthreads();
  }

  float* cp = ctx + (size_t)bh * M_ * DH_;
  #pragma unroll
  for (int mt = 0; mt < 2; ++mt)
    #pragma unroll
    for (int et = 0; et < 4; ++et)
      #pragma unroll
      for (int vv = 0; vv < 8; ++vv)
        cp[(wave * 32 + mt * 16 + vv + 8 * hi) * DH_ + et * 16 + lr] =
            acc[mt][et][vv];
  if (lr == 0) {
    #pragma unroll
    for (int mt = 0; mt < 2; ++mt)
      #pragma unroll
      for (int vv = 0; vv < 8; ++vv)
        ksum[bh * M_ + wave * 32 + mt * 16 + vv + 8 * hi] = acc[mt][4][vv];
  }
}

// ---------------------------------------------------------------------------
// Phase 2: per (b,h, 64 q-rows): qp features, out = (qp @ ctx) * d_inv.
// The per-(b,h) global max of dp_q cancels exactly in num/den -> omitted.
// ---------------------------------------------------------------------------
__global__ __launch_bounds__(256)
void performer_phase2(const float* __restrict__ qg, const float* __restrict__ proj,
                      const float* __restrict__ ctx, const float* __restrict__ ksum,
                      float* __restrict__ xout) {
  __shared__ _Float16 projB[M_][DH_ + 8];   // col-major B: [m][d]
  __shared__ _Float16 ctxT[DH_][M_ + 8];    // col-major B: [e][k]
  __shared__ _Float16 qph[64][M_ + 8];      // A: [n][m]
  __shared__ _Float16 qh[64][DH_ + 8];      // A: [n][d] (scaled)
  __shared__ float ksl[M_];
  __shared__ float dnv[64];
  __shared__ float dinv[64];
  __shared__ float dnpart[64][4];
  __shared__ float denpart[64][4];

  const int bh = blockIdx.y;
  const int b = bh / H_, h = bh % H_;
  const int n0 = blockIdx.x * 64;
  const int tid = threadIdx.x;
  const int wave = tid >> 5, lane = tid & 31, lr = lane & 15, hi = lane >> 4;

  {
    const float* cb = ctx + (size_t)bh * M_ * DH_;
    for (int i = tid; i < (M_ * DH_) / 4; i += 256) {
      const float4 f = *(const float4*)(proj + i * 4);
      v4h h4; h4[0] = (_Float16)f.x; h4[1] = (_Float16)f.y;
      h4[2] = (_Float16)f.z; h4[3] = (_Float16)f.w;
      *(v4h*)&projB[(i * 4) >> 6][(i * 4) & 63] = h4;
      const float4 g = *(const float4*)(cb + i * 4);
      const int kkk = (i * 4) >> 6, e0 = (i * 4) & 63;
      ctxT[e0 + 0][kkk] = (_Float16)g.x;
      ctxT[e0 + 1][kkk] = (_Float16)g.y;
      ctxT[e0 + 2][kkk] = (_Float16)g.z;
      ctxT[e0 + 3][kkk] = (_Float16)g.w;
    }
  }
  ksl[tid] = ksum[bh * M_ + tid];

  // q tile load + diag partials
  {
    const int row = tid >> 2, off = (tid & 3) * 16;
    const size_t base = ((size_t)b * N_ + n0 + row) * D_ + h * DH_ + off;
    const float4 f0 = *(const float4*)(qg + base);
    const float4 f1 = *(const float4*)(qg + base + 4);
    const float4 f2 = *(const float4*)(qg + base + 8);
    const float4 f3 = *(const float4*)(qg + base + 12);
    dnpart[row][tid & 3] =
        f0.x * f0.x + f0.y * f0.y + f0.z * f0.z + f0.w * f0.w +
        f1.x * f1.x + f1.y * f1.y + f1.z * f1.z + f1.w * f1.w +
        f2.x * f2.x + f2.y * f2.y + f2.z * f2.z + f2.w * f2.w +
        f3.x * f3.x + f3.y * f3.y + f3.z * f3.z + f3.w * f3.w;
    *(v8h*)&qh[row][off]     = cvt8(f0, f1, DATA_SCALE);
    *(v8h*)&qh[row][off + 8] = cvt8(f2, f3, DATA_SCALE);
  }
  __syncthreads();

  if (tid < 64) {
    const float s = dnpart[tid][0] + dnpart[tid][1] + dnpart[tid][2] + dnpart[tid][3];
    dnv[tid] = s * (0.5f * DATA_SCALE * DATA_SCALE);
  }
  __syncthreads();

  // ---- GEMM1: dp[64x256] = qh @ projB^T, exp on D-fragments -> qph --------
  #pragma unroll
  for (int i = 0; i < 8; ++i) {
    const int t = i * 8 + wave;            // 4 n-tiles x 16 m-tiles
    const int ntile = t >> 4, mtile = t & 15;
    const _Float16* ap = &qh[ntile * 16 + lr][0];
    const _Float16* bp = &projB[mtile * 16 + lr][0];
    v8f c;
    #pragma unroll
    for (int vv = 0; vv < 8; ++vv) c[vv] = 0.f;
    c = wmma_f16f32(fragA(ap, 0, hi), fragB(bp, 0, hi), c);
    c = wmma_f16f32(fragA(ap, 32, hi), fragB(bp, 32, hi), c);
    #pragma unroll
    for (int vv = 0; vv < 8; ++vv) {
      const int n = ntile * 16 + vv + 8 * hi;
      qph[n][mtile * 16 + lr] =
          (_Float16)(SCALE_ * __expf(c[vv] - dnv[n] + EPS_));
    }
  }
  __syncthreads();

  // ---- denominator: d_inv[n] = 1 / (qp[n,:] . ksum) (parallel partials) ---
  {
    const int rr = tid >> 2, seg = (tid & 3) * 64;
    float s = 0.f;
    #pragma unroll
    for (int m = 0; m < 64; ++m) s += (float)qph[rr][seg + m] * ksl[seg + m];
    denpart[rr][tid & 3] = s;
  }
  __syncthreads();
  if (tid < 64)
    dinv[tid] = 1.f / (denpart[tid][0] + denpart[tid][1] +
                       denpart[tid][2] + denpart[tid][3]);
  __syncthreads();

  // ---- GEMM2: out[64x64] = qp @ ctx  (K=256, 8 k-steps) -------------------
  #pragma unroll
  for (int i = 0; i < 2; ++i) {
    const int t = i * 8 + wave;            // 4 n-tiles x 4 e-tiles
    const int ntile = t >> 2, et = t & 3;
    const _Float16* ap = &qph[ntile * 16 + lr][0];
    const _Float16* bp = &ctxT[et * 16 + lr][0];
    v8f c;
    #pragma unroll
    for (int vv = 0; vv < 8; ++vv) c[vv] = 0.f;
    #pragma unroll
    for (int ks = 0; ks < 8; ++ks)
      c = wmma_f16f32(fragA(ap, ks * 32, hi), fragB(bp, ks * 32, hi), c);
    #pragma unroll
    for (int vv = 0; vv < 8; ++vv) {
      const int n = ntile * 16 + vv + 8 * hi;
      xout[((size_t)b * N_ + n0 + n) * D_ + h * DH_ + et * 16 + lr] =
          c[vv] * dinv[n];
    }
  }
}

// ---------------------------------------------------------------------------
// Phase 3: Y = X @ W^T + bias  ([16384 x 1024] @ [1024 x 1024]^T)
// W is [out][in] row-major == col-major-by-out for the B operand: direct copy.
// ---------------------------------------------------------------------------
__global__ __launch_bounds__(256)
void performer_out_linear(const float* __restrict__ X, const float* __restrict__ W,
                          const float* __restrict__ bias, float* __restrict__ Y) {
  __shared__ _Float16 xs[64][64 + 8];    // A: [r][i]
  __shared__ _Float16 wB[64][64 + 8];    // col-major B: [o][i]
  const int o0 = blockIdx.x * 64;
  const int r0 = blockIdx.y * 64;
  const int tid = threadIdx.x;
  const int wave = tid >> 5, lane = tid & 31, lr = lane & 15, hi = lane >> 4;

  v8f acc[2];
  #pragma unroll
  for (int i = 0; i < 2; ++i)
    #pragma unroll
    for (int vv = 0; vv < 8; ++vv) acc[i][vv] = 0.f;

  const int row = tid >> 2, off = (tid & 3) * 16;

  for (int kk = 0; kk < D_; kk += 64) {
    {
      const float* xp = X + (size_t)(r0 + row) * D_ + kk + off;
      const float* wp = W + (size_t)(o0 + row) * D_ + kk + off;
      const float4 x0 = *(const float4*)(xp);
      const float4 x1 = *(const float4*)(xp + 4);
      const float4 x2 = *(const float4*)(xp + 8);
      const float4 x3 = *(const float4*)(xp + 12);
      *(v8h*)&xs[row][off]     = cvt8(x0, x1, 1.f);
      *(v8h*)&xs[row][off + 8] = cvt8(x2, x3, 1.f);
      const float4 w0 = *(const float4*)(wp);
      const float4 w1 = *(const float4*)(wp + 4);
      const float4 w2 = *(const float4*)(wp + 8);
      const float4 w3 = *(const float4*)(wp + 12);
      *(v8h*)&wB[row][off]     = cvt8(w0, w1, 1.f);
      *(v8h*)&wB[row][off + 8] = cvt8(w2, w3, 1.f);
    }
    __syncthreads();
    #pragma unroll
    for (int i = 0; i < 2; ++i) {
      const int t = i * 8 + wave;          // 4 r-tiles x 4 o-tiles
      const int rt = t >> 2, ot = t & 3;
      const _Float16* ap = &xs[rt * 16 + lr][0];
      const _Float16* bp = &wB[ot * 16 + lr][0];
      #pragma unroll
      for (int ks = 0; ks < 2; ++ks)
        acc[i] = wmma_f16f32(fragA(ap, ks * 32, hi), fragB(bp, ks * 32, hi),
                             acc[i]);
    }
    __syncthreads();
  }

  #pragma unroll
  for (int i = 0; i < 2; ++i) {
    const int t = i * 8 + wave;
    const int rt = t >> 2, ot = t & 3;
    const float bcol = bias[o0 + ot * 16 + lr];
    #pragma unroll
    for (int vv = 0; vv < 8; ++vv)
      Y[(size_t)(r0 + rt * 16 + vv + 8 * hi) * D_ + o0 + ot * 16 + lr] =
          acc[i][vv] + bcol;
  }
}

// ---------------------------------------------------------------------------
extern "C" void kernel_launch(void* const* d_in, const int* in_sizes, int n_in,
                              void* d_out, int out_size, void* d_ws, size_t ws_size,
                              hipStream_t stream) {
  const float* q    = (const float*)d_in[0];
  const float* k    = (const float*)d_in[1];
  const float* v    = (const float*)d_in[2];
  const float* W    = (const float*)d_in[3];
  const float* bias = (const float*)d_in[4];
  const float* proj = (const float*)d_in[5];
  float* out = (float*)d_out;

  float* ctx  = (float*)d_ws;                                    // [B,H,M,DH]
  float* ksum = ctx + (size_t)B_ * H_ * M_ * DH_;                 // [B,H,M]
  float* X    = ksum + (size_t)B_ * H_ * M_;                      // [B,N,D]

  performer_phase1<<<dim3(B_ * H_), dim3(256), 0, stream>>>(k, v, proj, ctx, ksum);
  performer_phase2<<<dim3(N_ / 64, B_ * H_), dim3(256), 0, stream>>>(q, proj, ctx,
                                                                     ksum, X);
  performer_out_linear<<<dim3(D_ / 64, (B_ * N_) / 64), dim3(256), 0, stream>>>(
      X, W, bias, out);
}
